// StateActionEncoder_64218351009906
// MI455X (gfx1250) — compile-verified, matrix operand order checked by reading, not run on hardware
//
#include <hip/hip_runtime.h>
#include <hip/hip_bf16.h>

// ---------------------------------------------------------------------------
// Relational state/action encoder for MI455X (gfx1250).
// All matmuls via v_wmma_f32_16x16x32_bf16 (fp32 accumulate); weights
// pre-transposed to bf16; concat-GEMMs algebraically split so i/j-independent
// halves run on 2048 rows and are broadcast-added in 65536-row GEMM epilogues.
// GEMM: 128x64 block tile, 4 waves (each 32Mx64N), K-step 32, double-buffered
// LDS fed by gfx1250 async global->LDS copies (ASYNCcnt) when available.
// ---------------------------------------------------------------------------

typedef unsigned short bhalf;                                  // raw bf16 bits
typedef __attribute__((ext_vector_type(16))) __bf16 v16bf;
typedef __attribute__((ext_vector_type(8)))  float  v8f;
typedef __attribute__((ext_vector_type(4)))  int    v4i;

#if defined(__has_builtin)
#if __has_builtin(__builtin_amdgcn_global_load_async_to_lds_b128)
#define ASYNC_COPY 1
#endif
#endif
#ifndef ASYNC_COPY
#define ASYNC_COPY 0
#endif

#if ASYNC_COPY
#if __has_builtin(__builtin_amdgcn_s_wait_asynccnt)
#define WAIT_ASYNC() __builtin_amdgcn_s_wait_asynccnt(0)
#else
#define WAIT_ASYNC() asm volatile("s_wait_asynccnt 0x0" ::: "memory")
#endif
static __device__ __forceinline__ void async_cp16(const bhalf* g, bhalf* l) {
  __builtin_amdgcn_global_load_async_to_lds_b128((v4i*)(void*)g, (v4i*)(void*)l, 0, 0);
}
#endif

static __device__ __forceinline__ bhalf f2bf(float f) {
  unsigned int u = __builtin_bit_cast(unsigned int, f);
  u = u + 0x7FFFu + ((u >> 16) & 1u);                          // round-nearest-even
  return (bhalf)(u >> 16);
}
static __device__ __forceinline__ float bf2f(bhalf h) {
  unsigned int u = ((unsigned int)h) << 16;
  return __builtin_bit_cast(float, u);
}

// ---- weight convert + transpose: W[K][N] fp32 -> Wt[N][K] bf16 -------------
__global__ void k_wt(const float* __restrict__ W, bhalf* __restrict__ Wt,
                     int K, int N) {
  int idx = blockIdx.x * 256 + threadIdx.x;
  if (idx >= K * N) return;
  int k = idx / N, n = idx % N;
  Wt[(size_t)n * K + k] = f2bf(W[idx]);
}

// ---- elementwise fp32 -> bf16 ---------------------------------------------
__global__ void k_cv(const float* __restrict__ x, bhalf* __restrict__ y, int n) {
  int idx = blockIdx.x * 256 + threadIdx.x;
  if (idx < n) y[idx] = f2bf(x[idx]);
}

// ---- generic bf16 WMMA GEMM -----------------------------------------------
// C[M,N] = relu?( A[M,K](bf16,lda) * Bt[N,K](bf16,ldb)^T + bias
//                 + addmode>=1 ? T1[b*32+j, n] : 0
//                 + addmode==2 ? T2[b*32+i, n] : 0 )
// rho = b*1024 + i*32 + j. M mult of 128, N mult of 64, K mult of 32.
// Block: 128 threads (4 waves); block tile 128(M)x64(N); wave tile 32x64.
#define PITCH 40   // bf16 elems per LDS tile row: 16B-aligned, bank-spread

__global__ __launch_bounds__(128)
void k_gemm(const bhalf* __restrict__ A, int lda,
            const bhalf* __restrict__ Bt, int ldb,
            const float* __restrict__ bias,
            const float* __restrict__ T1, const float* __restrict__ T2,
            int addmode,
            float* __restrict__ outF, bhalf* __restrict__ outBF,
            int N, int K, int do_relu) {
  __shared__ bhalf At[2][128 * PITCH];           // 2 x 10240 B
  __shared__ bhalf Bl[2][ 64 * PITCH];           // 2 x  5120 B

  const int  tid  = threadIdx.x;
  const int  wave = tid >> 5;
  const int  lane = tid & 31;
  const long m0   = (long)blockIdx.x * 128;
  const int  n0   = blockIdx.y * 64;

  // per-thread copy chunks: c = tid + i*128, i<6; c<512 -> A-tile, else B-tile.
  // (branches are uniform after unrolling: i<4 always A, i>=4 always B)
  auto g_src = [&](int c, int k0) -> const bhalf* {
    if (c < 512) { int row = c >> 2, part = c & 3;
                   return A + (size_t)(m0 + row) * lda + k0 + part * 8; }
    int cc = c - 512; int row = cc >> 2, part = cc & 3;
    return Bt + (size_t)(n0 + row) * ldb + k0 + part * 8;
  };
  auto l_dst = [&](int c, int buf) -> bhalf* {
    if (c < 512) { int row = c >> 2, part = c & 3;
                   return &At[buf][row * PITCH + part * 8]; }
    int cc = c - 512; int row = cc >> 2, part = cc & 3;
    return &Bl[buf][row * PITCH + part * 8];
  };

  const v8f zero = {0.f, 0.f, 0.f, 0.f, 0.f, 0.f, 0.f, 0.f};
  v8f acc[2][4] = {{zero, zero, zero, zero}, {zero, zero, zero, zero}};

  const int fm   = lane & 15;                    // fragment row/col within 16
  const int koff = (lane >> 4) << 3;             // K-half select per ISA layout
  const int nk   = K >> 5;

  union Frag { v16bf v; uint4 q[2]; };

  // ---- preload tile 0 into buffer 0 ----
#if ASYNC_COPY
  #pragma unroll
  for (int i = 0; i < 6; ++i) async_cp16(g_src(tid + i * 128, 0), l_dst(tid + i * 128, 0));
  WAIT_ASYNC();
#else
  {
    uint4 r[6];
    #pragma unroll
    for (int i = 0; i < 6; ++i) r[i] = *(const uint4*)g_src(tid + i * 128, 0);
    #pragma unroll
    for (int i = 0; i < 6; ++i) *(uint4*)l_dst(tid + i * 128, 0) = r[i];
  }
#endif
  __syncthreads();

  int cur = 0;
  for (int t = 0; t < nk; ++t) {
    const int  nxt      = cur ^ 1;
    const bool has_next = (t + 1 < nk);

#if ASYNC_COPY
    if (has_next) {
      #pragma unroll
      for (int i = 0; i < 6; ++i)
        async_cp16(g_src(tid + i * 128, (t + 1) << 5), l_dst(tid + i * 128, nxt));
    }
#else
    uint4 r[6];
    if (has_next) {
      #pragma unroll
      for (int i = 0; i < 6; ++i) r[i] = *(const uint4*)g_src(tid + i * 128, (t + 1) << 5);
    }
#endif

    // ---- compute on buffer `cur`: 2 A-frags x 4 B-frags = 8 WMMA ----
    Frag fa0, fa1;
    const bhalf* ap = &At[cur][(wave * 32 + fm) * PITCH];
    fa0.q[0] = *(const uint4*)(ap + koff);
    fa0.q[1] = *(const uint4*)(ap + 16 + koff);
    fa1.q[0] = *(const uint4*)(ap + 16 * PITCH + koff);
    fa1.q[1] = *(const uint4*)(ap + 16 * PITCH + 16 + koff);
    #pragma unroll
    for (int j = 0; j < 4; ++j) {
      Frag fb;
      const bhalf* bp = &Bl[cur][(j * 16 + fm) * PITCH];
      fb.q[0] = *(const uint4*)(bp + koff);
      fb.q[1] = *(const uint4*)(bp + 16 + koff);
      acc[0][j] = __builtin_amdgcn_wmma_f32_16x16x32_bf16(
          false, fa0.v, false, fb.v, (short)0, acc[0][j], false, false);
      acc[1][j] = __builtin_amdgcn_wmma_f32_16x16x32_bf16(
          false, fa1.v, false, fb.v, (short)0, acc[1][j], false, false);
    }

#if ASYNC_COPY
    WAIT_ASYNC();                                // this wave's nxt-tile copies done
#else
    if (has_next) {
      #pragma unroll
      for (int i = 0; i < 6; ++i) *(uint4*)l_dst(tid + i * 128, nxt) = r[i];
    }
#endif
    __syncthreads();
    cur = nxt;
  }

  // epilogue; C/D layout: elem e -> row = e + 8*(lane>>4), col = lane&15
  const int crow0 = (lane >> 4) << 3;
  const int ccol  = lane & 15;
  #pragma unroll
  for (int sub = 0; sub < 2; ++sub) {
    #pragma unroll
    for (int j = 0; j < 4; ++j) {
      const int nG = n0 + j * 16 + ccol;
      const float bv = bias ? bias[nG] : 0.f;
      #pragma unroll
      for (int e = 0; e < 8; ++e) {
        const long row = m0 + wave * 32 + sub * 16 + crow0 + e;
        float v = acc[sub][j][e] + bv;
        if (addmode >= 1) {
          const long bj = ((row >> 10) << 5) | (row & 31);   // b*32 + j
          v += T1[bj * 256 + nG];
          if (addmode == 2) v += T2[(row >> 5) * 256 + nG];  // b*32 + i
        }
        if (do_relu) v = fmaxf(v, 0.f);
        if (outF)  outF[row * (long)N + nG]  = v;
        if (outBF) outBF[row * (long)N + nG] = f2bf(v);
      }
    }
  }
}

// ---- masked mean/max neighbor reduction + aggregation-input concat ---------
// r: [2048*32, 512] bf16 ; s_bf: [2048,256] bf16 ; cat: [2048, 768] bf16
__global__ void k_redcat(const bhalf* __restrict__ r,
                         const bhalf* __restrict__ s_bf,
                         const int* __restrict__ alive,
                         bhalf* __restrict__ cat) {
  const int bi = blockIdx.x;                    // b*32 + i
  const int h  = threadIdx.x;                   // 0..255
  const int b  = bi >> 5;
  const bhalf* base = r + ((size_t)bi << 5) * 512;
  float sum = 0.f, mx = -3.4e38f;
  #pragma unroll 4
  for (int j = 0; j < 32; ++j) {
    const float mk = (float)alive[(b << 5) + j];
    sum += bf2f(base[(size_t)j * 512 + h]) * mk;
    mx = fmaxf(mx, bf2f(base[(size_t)j * 512 + 256 + h]) * mk);
  }
  const size_t o = (size_t)bi * 768;
  cat[o + h]       = s_bf[(size_t)bi * 256 + h];
  cat[o + 256 + h] = f2bf(sum * (1.f / 32.f));
  cat[o + 512 + h] = f2bf(mx);
}

// ---------------------------------------------------------------------------
extern "C" void kernel_launch(void* const* d_in, const int* in_sizes, int n_in,
                              void* d_out, int out_size, void* d_ws, size_t ws_size,
                              hipStream_t stream) {
  (void)in_sizes; (void)n_in; (void)out_size; (void)ws_size;

  const float* states    = (const float*)d_in[0];   // [2048,256]
  const float* relations = (const float*)d_in[1];   // [65536,64]
  const int*   alive     = (const int*)  d_in[2];   // [2048]
  const float* se_w1 = (const float*)d_in[3];  const float* se_b1 = (const float*)d_in[4];
  const float* se_w2 = (const float*)d_in[5];  const float* se_b2 = (const float*)d_in[6];
  const float* re_w1 = (const float*)d_in[7];  const float* re_b1 = (const float*)d_in[8];
  const float* re_w2 = (const float*)d_in[9];  const float* re_b2 = (const float*)d_in[10];
  const float* ag_w  = (const float*)d_in[11]; const float* ag_b  = (const float*)d_in[12];
  const float* ae_w1 = (const float*)d_in[13]; const float* ae_b1 = (const float*)d_in[14];
  const float* ae_w2 = (const float*)d_in[15]; const float* ae_b2 = (const float*)d_in[16];

  char* ws = (char*)d_ws;
  // bf16 transposed weights
  bhalf* se_w1t = (bhalf*)(ws + 0);          // 256x256 -> 131072 B
  bhalf* se_w2t = (bhalf*)(ws + 131072);     // 256x256
  bhalf* re_w1t = (bhalf*)(ws + 262144);     // [256][320] -> 163840 B
  bhalf* re_w2t = (bhalf*)(ws + 425984);     // [512][256] -> 262144 B
  bhalf* ag_wt  = (bhalf*)(ws + 688128);     // [256][768] -> 393216 B
  bhalf* ae_w1t = (bhalf*)(ws + 1081344);    // [256][1024] -> 524288 B
  bhalf* ae_w2t = (bhalf*)(ws + 1605632);    // [512][256] -> 262144 B
  // activations / intermediates
  bhalf* x_bf    = (bhalf*)(ws + 2097152);   // states bf16      1 MB
  bhalf* hse_bf  = (bhalf*)(ws + 3145728);   // se hidden / reused as state_bf, 1 MB
  bhalf* s_bf    = (bhalf*)(ws + 4194304);   // s bf16           1 MB
  float* Yf      = (float*)(ws + 5242880);   // s_j@re_w1 + b1   2 MB
  bhalf* rel_bf  = (bhalf*)(ws + 14680064);  // relations bf16   8 MB (dead after re1)
  bhalf* cat_bf  = (bhalf*)(ws + 14680064);  // overlaps rel_bf  3 MB
  float* ZJf     = (float*)(ws + 17825792);  // overlaps rel_bf  2 MB
  float* ZIf     = (float*)(ws + 19922944);  // overlaps rel_bf  2 MB
  bhalf* h_big   = (bhalf*)(ws + 23068672);  // [65536,256] bf16 32 MB (re/ae hidden)
  bhalf* r_buf   = (bhalf*)(ws + 56623104);  // [65536,512] bf16 64 MB

  float* out_state = (float*)d_out;                 // [2048,256]
  float* out_act   = (float*)d_out + 2048 * 256;    // [65536,512]

  auto gemm = [&](const bhalf* A, int lda, const bhalf* Bt, int ldb,
                  const float* bias, const float* T1, const float* T2, int addmode,
                  float* oF, bhalf* oBF, int M, int N, int K, int relu) {
    dim3 g(M / 128, N / 64);
    k_gemm<<<g, 128, 0, stream>>>(A, lda, Bt, ldb, bias, T1, T2, addmode,
                                  oF, oBF, N, K, relu);
  };

  // weight prep
  k_wt<<<(256 * 256 + 255) / 256, 256, 0, stream>>>(se_w1, se_w1t, 256, 256);
  k_wt<<<(256 * 256 + 255) / 256, 256, 0, stream>>>(se_w2, se_w2t, 256, 256);
  k_wt<<<(320 * 256 + 255) / 256, 256, 0, stream>>>(re_w1, re_w1t, 320, 256);
  k_wt<<<(256 * 512 + 255) / 256, 256, 0, stream>>>(re_w2, re_w2t, 256, 512);
  k_wt<<<(768 * 256 + 255) / 256, 256, 0, stream>>>(ag_w,  ag_wt,  768, 256);
  k_wt<<<(1024 * 256 + 255) / 256, 256, 0, stream>>>(ae_w1, ae_w1t, 1024, 256);
  k_wt<<<(256 * 512 + 255) / 256, 256, 0, stream>>>(ae_w2, ae_w2t, 256, 512);
  k_cv<<<(2048 * 256 + 255) / 256, 256, 0, stream>>>(states, x_bf, 2048 * 256);
  k_cv<<<(65536 * 64 + 255) / 256, 256, 0, stream>>>(relations, rel_bf, 65536 * 64);

  // state MLP: s = relu(relu(x@W1+b1)@W2+b2)
  gemm(x_bf, 256, se_w1t, 256, se_b1, nullptr, nullptr, 0, nullptr, hse_bf, 2048, 256, 256, 1);
  gemm(hse_bf, 256, se_w2t, 256, se_b2, nullptr, nullptr, 0, nullptr, s_bf, 2048, 256, 256, 1);

  // Y[b,j] = s[b,j] @ re_w1[64:,:] + re_b1   (i-independent half of relation layer 1)
  gemm(s_bf, 256, re_w1t + 64, 320, re_b1, nullptr, nullptr, 0, Yf, nullptr, 2048, 256, 256, 0);
  // relation layer 1: h = relu(relations @ re_w1[:64,:] + Y[b,j])
  gemm(rel_bf, 64, re_w1t, 320, nullptr, Yf, nullptr, 1, nullptr, h_big, 65536, 256, 64, 1);
  // relation layer 2: r = relu(h @ re_w2 + b2)
  gemm(h_big, 256, re_w2t, 256, re_b2, nullptr, nullptr, 0, nullptr, r_buf, 65536, 512, 256, 1);

  // masked mean/max over neighbors + concat [s, avg, max]
  k_redcat<<<2048, 256, 0, stream>>>(r_buf, s_bf, alive, cat_bf);

  // aggregation: state = relu(cat @ ag_w + ag_b)  (fp32 output + bf16 copy)
  gemm(cat_bf, 768, ag_wt, 768, ag_b, nullptr, nullptr, 0, out_state, hse_bf, 2048, 256, 768, 1);

  // i/j-independent halves of action layer 1
  gemm(hse_bf, 256, ae_w1t + 512, 1024, ae_b1, nullptr, nullptr, 0, ZJf, nullptr, 2048, 256, 256, 0);
  gemm(hse_bf, 256, ae_w1t + 768, 1024, nullptr, nullptr, nullptr, 0, ZIf, nullptr, 2048, 256, 256, 0);
  // action layer 1: h = relu(r @ ae_w1[:512,:] + ZJ[b,j] + ZI[b,i])
  gemm(r_buf, 512, ae_w1t, 1024, nullptr, ZJf, ZIf, 2, nullptr, h_big, 65536, 256, 512, 1);
  // action layer 2: a = relu(h @ ae_w2 + b2)  -> fp32 output
  gemm(h_big, 256, ae_w2t, 256, ae_b2, nullptr, nullptr, 0, out_act, nullptr, 65536, 512, 256, 1);
}
